// PointNetFeaturePropagation_72129680769837
// MI455X (gfx1250) — compile-verified, matrix operand order checked by reading, not run on hardware
//
#include <hip/hip_runtime.h>

// ---------------- problem constants ----------------
#define BB   8
#define NN   8192
#define SS   2048
#define D1   128
#define D2   256
#define C0   384          // IN_CH = D1 + D2
#define H0   256          // MLP[0]
#define H1   128          // MLP[1]
#define MTOT (BB * NN)    // 65536 rows
#define BN_EPS 1e-5f

typedef __attribute__((ext_vector_type(16))) _Float16 v16h;
typedef __attribute__((ext_vector_type(8)))  _Float16 v8h;
typedef __attribute__((ext_vector_type(2)))  _Float16 v2h;
typedef __attribute__((ext_vector_type(8)))  float    v8f;

// ---------------------------------------------------------------------------
// Load a 16x32 f16 WMMA fragment (A operand layout; B uses same layout on the
// transposed matrix). Per lane: row = base + (lane&15); K-halves at
// kh = (lane>>4)*8 : two contiguous 16-byte chunks [kh..kh+7] and [kh+16..kh+23].
// ---------------------------------------------------------------------------
__device__ __forceinline__ v16h ldfrag(const _Float16* __restrict__ p) {
    v8h lo = *(const v8h*)(p);
    v8h hi = *(const v8h*)(p + 16);
    v16h r;
#pragma unroll
    for (int i = 0; i < 8; ++i) { r[i] = lo[i]; r[i + 8] = hi[i]; }
    return r;
}
__device__ __forceinline__ v16h ldfrag_lds(const _Float16* p) {
    v8h lo = *(const v8h*)(p);
    v8h hi = *(const v8h*)(p + 16);
    v16h r;
#pragma unroll
    for (int i = 0; i < 8; ++i) { r[i] = lo[i]; r[i + 8] = hi[i]; }
    return r;
}

// ---------------------------------------------------------------------------
// prep: cast weights to f16, zero BN-stat accumulators
// ---------------------------------------------------------------------------
__global__ __launch_bounds__(256) void prep_kernel(
    const float* __restrict__ w0, const float* __restrict__ w1,
    _Float16* __restrict__ w0h, _Float16* __restrict__ w1h,
    float* __restrict__ st0, float* __restrict__ st1)
{
    int i = blockIdx.x * 256 + threadIdx.x;
    if (i < H0 * C0) w0h[i] = (_Float16)w0[i];
    if (i < H1 * H0) w1h[i] = (_Float16)w1[i];
    if (i < 2 * H0)  st0[i] = 0.0f;
    if (i < 2 * H1)  st1[i] = 0.0f;
}

// ---------------------------------------------------------------------------
// transpose points2 (B, D2, S) -> featsT (B, S, D2), f32, LDS tiled
// grid: (S/32, D2/32, B), block 256 = (32 lanes x 8 rows)
// ---------------------------------------------------------------------------
__global__ __launch_bounds__(256) void transpose_p2_kernel(
    const float* __restrict__ p2, float* __restrict__ ft)
{
    __shared__ float tile[32][33];
    const int b  = blockIdx.z;
    const int s0 = blockIdx.x * 32;
    const int d0 = blockIdx.y * 32;
    const int tx = threadIdx.x & 31, ty = threadIdx.x >> 5;
#pragma unroll
    for (int j = 0; j < 32; j += 8)
        tile[ty + j][tx] = p2[((size_t)b * D2 + d0 + ty + j) * SS + s0 + tx];
    __syncthreads();
#pragma unroll
    for (int j = 0; j < 32; j += 8)
        ft[((size_t)b * SS + s0 + ty + j) * D2 + d0 + tx] = tile[tx][ty + j];
}

// ---------------------------------------------------------------------------
// points1 (B, D1, N) -> x0h columns [0,128) of row-major (M, 384) f16
// grid: (N/32, D1/32, B), block 256
// ---------------------------------------------------------------------------
__global__ __launch_bounds__(256) void p1_to_x0_kernel(
    const float* __restrict__ p1, _Float16* __restrict__ x0h)
{
    __shared__ float tile[32][33];
    const int b  = blockIdx.z;
    const int n0 = blockIdx.x * 32;
    const int c0 = blockIdx.y * 32;
    const int tx = threadIdx.x & 31, ty = threadIdx.x >> 5;
#pragma unroll
    for (int j = 0; j < 32; j += 8)
        tile[ty + j][tx] = p1[((size_t)b * D1 + c0 + ty + j) * NN + n0 + tx];
    __syncthreads();
#pragma unroll
    for (int j = 0; j < 32; j += 8)
        x0h[((size_t)b * NN + n0 + ty + j) * C0 + c0 + tx] = (_Float16)tile[tx][ty + j];
}

// ---------------------------------------------------------------------------
// 3-NN + inverse-distance weights. xyz2 staged in LDS (24 KB).
// grid: (N/256, B), block 256, one query point per thread.
// ---------------------------------------------------------------------------
__global__ __launch_bounds__(256) void knn_kernel(
    const float* __restrict__ xyz1, const float* __restrict__ xyz2,
    int* __restrict__ idx4, float* __restrict__ w4)
{
    __shared__ float sx[SS], sy[SS], sz[SS];
    const int b = blockIdx.y;
    const int n = blockIdx.x * 256 + threadIdx.x;
    const float* x2 = xyz2 + (size_t)b * 3 * SS;
    for (int i = threadIdx.x; i < SS; i += 256) {
        sx[i] = x2[i]; sy[i] = x2[SS + i]; sz[i] = x2[2 * SS + i];
    }
    __syncthreads();
    const float* x1 = xyz1 + (size_t)b * 3 * NN;
    const float px = x1[n], py = x1[NN + n], pz = x1[2 * NN + n];

    float d0 = 3.0e38f, d1 = 3.0e38f, d2 = 3.0e38f;
    int   i0 = 0, i1 = 0, i2 = 0;
    for (int s = 0; s < SS; ++s) {
        const float dx = px - sx[s], dy = py - sy[s], dz = pz - sz[s];
        const float d = dx * dx + dy * dy + dz * dz;
        if (d < d2) {
            if (d < d1) {
                if (d < d0) { d2 = d1; i2 = i1; d1 = d0; i1 = i0; d0 = d; i0 = s; }
                else        { d2 = d1; i2 = i1; d1 = d;  i1 = s; }
            } else          { d2 = d;  i2 = s; }
        }
    }
    const float r0 = 1.0f / (d0 + 1e-8f);
    const float r1 = 1.0f / (d1 + 1e-8f);
    const float r2 = 1.0f / (d2 + 1e-8f);
    const float rs = 1.0f / (r0 + r1 + r2);
    const size_t m = (size_t)b * NN + n;
    *(int4*)  (idx4 + m * 4) = make_int4(i0, i1, i2, 0);
    *(float4*)(w4   + m * 4) = make_float4(r0 * rs, r1 * rs, r2 * rs, 0.0f);
}

// ---------------------------------------------------------------------------
// interpolation: x0h columns [128, 384) = sum_k w_k * featsT[b, idx_k, :]
// grid: MTOT blocks of 128 threads, 2 channels per thread (float2 reads).
// ---------------------------------------------------------------------------
__global__ __launch_bounds__(128) void interp_kernel(
    const float* __restrict__ featsT, const int* __restrict__ idx4,
    const float* __restrict__ w4, _Float16* __restrict__ x0h)
{
    const int m = blockIdx.x;
    const int b = m >> 13;           // / NN
    const int4   id = *(const int4*)  (idx4 + (size_t)m * 4);
    const float4 w  = *(const float4*)(w4   + (size_t)m * 4);
    const float* f0 = featsT + ((size_t)b * SS + id.x) * D2;
    const float* f1 = featsT + ((size_t)b * SS + id.y) * D2;
    const float* f2 = featsT + ((size_t)b * SS + id.z) * D2;
    const int d = threadIdx.x * 2;
    const float2 a0 = *(const float2*)(f0 + d);
    const float2 a1 = *(const float2*)(f1 + d);
    const float2 a2 = *(const float2*)(f2 + d);
    v2h h;
    h[0] = (_Float16)(w.x * a0.x + w.y * a1.x + w.z * a2.x);
    h[1] = (_Float16)(w.x * a0.y + w.y * a1.y + w.z * a2.y);
    *(v2h*)(x0h + (size_t)m * C0 + D1 + d) = h;
}

// ---------------------------------------------------------------------------
// WMMA GEMM:  Y[M x O] = A[M x K] * W^T   (W given row-major [O x K]), + bias
// block = 256 threads (8 waves); block tile 128x64; wave tile 32x32 (2x2 WMMA).
// Weight slab staged in LDS, row padded +8 halves to spread banks.
// ---------------------------------------------------------------------------
template <int K, int O>
__global__ __launch_bounds__(256) void gemm_wmma_kernel(
    const _Float16* __restrict__ A, const _Float16* __restrict__ W,
    const float* __restrict__ bias, float* __restrict__ Y)
{
    constexpr int LDSK = K + 8;
    __shared__ _Float16 Bs[64 * LDSK];

    const int tid  = threadIdx.x;
    const int wave = tid >> 5;
    const int lane = tid & 31;
    const int mblk = blockIdx.x * 128;
    const int oblk = blockIdx.y * 64;

    // cooperative 16B-chunk load of the 64 x K weight slab into LDS
    constexpr int CH = K / 8;
    for (int i = tid; i < 64 * CH; i += 256) {
        const int o = i / CH, kc = i % CH;
        *(v8h*)(&Bs[o * LDSK + kc * 8]) =
            *(const v8h*)(W + (size_t)(oblk + o) * K + kc * 8);
    }
    __syncthreads();

    const int mw   = mblk + (wave & 3) * 32;   // wave row origin (global)
    const int ow   = (wave >> 2) * 32;         // wave col origin (LDS-local)
    const int lrow = lane & 15;
    const int lkh  = (lane >> 4) * 8;

    const _Float16* A0 = A + (size_t)(mw + lrow) * K + lkh;
    const _Float16* A1 = A + (size_t)(mw + 16 + lrow) * K + lkh;
    const _Float16* B0 = &Bs[(ow + lrow) * LDSK + lkh];
    const _Float16* B1 = &Bs[(ow + 16 + lrow) * LDSK + lkh];

    v8f acc00 = {}, acc01 = {}, acc10 = {}, acc11 = {};

    for (int k0 = 0; k0 < K; k0 += 32) {
        const v16h a0 = ldfrag(A0 + k0);
        const v16h a1 = ldfrag(A1 + k0);
        const v16h b0 = ldfrag_lds(B0 + k0);
        const v16h b1 = ldfrag_lds(B1 + k0);
        acc00 = __builtin_amdgcn_wmma_f32_16x16x32_f16(false, a0, false, b0, (short)0, acc00, false, false);
        acc01 = __builtin_amdgcn_wmma_f32_16x16x32_f16(false, a0, false, b1, (short)0, acc01, false, false);
        acc10 = __builtin_amdgcn_wmma_f32_16x16x32_f16(false, a1, false, b0, (short)0, acc10, false, false);
        acc11 = __builtin_amdgcn_wmma_f32_16x16x32_f16(false, a1, false, b1, (short)0, acc11, false, false);
    }

    // C/D layout: lane -> col = oblk+ow+tj*16+(lane&15); vgpr r -> row m = (lane>>4)*8 + r
    const int rbase = (lane >> 4) * 8;
#pragma unroll
    for (int tj = 0; tj < 2; ++tj) {
        const int col = oblk + ow + tj * 16 + lrow;
        const float bv = bias[col];
        const v8f c0 = (tj == 0) ? acc00 : acc01;
        const v8f c1 = (tj == 0) ? acc10 : acc11;
#pragma unroll
        for (int r = 0; r < 8; ++r) {
            Y[(size_t)(mw + rbase + r) * O + col]      = c0[r] + bv;
            Y[(size_t)(mw + 16 + rbase + r) * O + col] = c1[r] + bv;
        }
    }
}

// ---------------------------------------------------------------------------
// BN stats: per-channel sum & sumsq over all M rows -> st[0..O)=sum, st[O..2O)=sumsq
// grid: MTOT/256 blocks, each reduces 256 rows; 2 f32 atomics per thread.
// ---------------------------------------------------------------------------
template <int O>
__global__ __launch_bounds__(256) void bn_stats_kernel(
    const float* __restrict__ Y, float* __restrict__ st)
{
    constexpr int RSTEP = 256 / O;
    const int o  = threadIdx.x % O;
    const int r0 = threadIdx.x / O;
    const size_t base = (size_t)blockIdx.x * 256;
    float s = 0.0f, q = 0.0f;
    for (int r = r0; r < 256; r += RSTEP) {
        const float v = Y[(base + r) * O + o];
        s += v; q += v * v;
    }
    atomicAdd(&st[o], s);
    atomicAdd(&st[O + o], q);
}

// ---------------------------------------------------------------------------
// BN apply + ReLU + f16 cast (layer-0 output -> layer-1 input)
// ---------------------------------------------------------------------------
template <int O>
__global__ __launch_bounds__(256) void bn_relu_cast_kernel(
    const float* __restrict__ Y, const float* __restrict__ st,
    const float* __restrict__ g, const float* __restrict__ be,
    _Float16* __restrict__ Xh)
{
    const size_t i = (size_t)blockIdx.x * 256 + threadIdx.x;
    const int o = (int)(i % O);
    const float cnt  = (float)MTOT;
    const float mean = st[o] / cnt;
    const float var  = st[O + o] / cnt - mean * mean;
    const float inv  = rsqrtf(var + BN_EPS);
    float v = (Y[i] - mean) * inv * g[o] + be[o];
    Xh[i] = (_Float16)fmaxf(v, 0.0f);
}

// ---------------------------------------------------------------------------
// Final BN + ReLU + transpose (M,128) row-major -> out (B, 128, N), LDS tiled
// grid: (N/32, H1/32, B), block 256
// ---------------------------------------------------------------------------
__global__ __launch_bounds__(256) void bn_relu_out_kernel(
    const float* __restrict__ Y, const float* __restrict__ st,
    const float* __restrict__ g, const float* __restrict__ be,
    float* __restrict__ out)
{
    __shared__ float tile[32][33];
    const int b  = blockIdx.z;
    const int n0 = blockIdx.x * 32;
    const int o0 = blockIdx.y * 32;
    const int tx = threadIdx.x & 31, ty = threadIdx.x >> 5;
    const float cnt = (float)MTOT;
    {
        const int o = o0 + tx;
        const float mean = st[o] / cnt;
        const float var  = st[H1 + o] / cnt - mean * mean;
        const float sc   = rsqrtf(var + BN_EPS) * g[o];
        const float sh   = be[o] - mean * sc;
#pragma unroll
        for (int j = 0; j < 32; j += 8) {
            const float v = Y[((size_t)b * NN + n0 + ty + j) * H1 + o] * sc + sh;
            tile[ty + j][tx] = fmaxf(v, 0.0f);
        }
    }
    __syncthreads();
#pragma unroll
    for (int j = 0; j < 32; j += 8)
        out[((size_t)b * H1 + o0 + ty + j) * NN + n0 + tx] = tile[tx][ty + j];
}

// ---------------------------------------------------------------------------
extern "C" void kernel_launch(void* const* d_in, const int* in_sizes, int n_in,
                              void* d_out, int out_size, void* d_ws, size_t ws_size,
                              hipStream_t stream)
{
    const float* xyz1 = (const float*)d_in[0];
    const float* xyz2 = (const float*)d_in[1];
    const float* pts1 = (const float*)d_in[2];
    const float* pts2 = (const float*)d_in[3];
    const float* w0   = (const float*)d_in[4];
    const float* b0   = (const float*)d_in[5];
    const float* g0   = (const float*)d_in[6];
    const float* be0  = (const float*)d_in[7];
    const float* w1   = (const float*)d_in[8];
    const float* b1   = (const float*)d_in[9];
    const float* g1   = (const float*)d_in[10];
    const float* be1  = (const float*)d_in[11];
    float* out = (float*)d_out;

    // workspace carve-up (256B aligned)
    char* base = (char*)d_ws;
    size_t off = 0;
    auto carve = [&](size_t bytes) -> char* {
        char* p = base + off;
        off += (bytes + 255) & ~(size_t)255;
        return p;
    };
    float*    featsT = (float*)   carve((size_t)BB * SS * D2 * 4);     // 16.8 MB
    _Float16* x0h    = (_Float16*)carve((size_t)MTOT * C0 * 2);        // 50.3 MB
    _Float16* w0h    = (_Float16*)carve((size_t)H0 * C0 * 2);
    _Float16* w1h    = (_Float16*)carve((size_t)H1 * H0 * 2);
    float*    y0     = (float*)   carve((size_t)MTOT * H0 * 4);        // 67 MB
    _Float16* x1h    = (_Float16*)carve((size_t)MTOT * H0 * 2);        // 33.5 MB
    float*    y1     = (float*)   carve((size_t)MTOT * H1 * 4);        // 33.5 MB
    int*      idx4   = (int*)     carve((size_t)MTOT * 4 * 4);
    float*    w4     = (float*)   carve((size_t)MTOT * 4 * 4);
    float*    st0    = (float*)   carve(2 * H0 * 4);
    float*    st1    = (float*)   carve(2 * H1 * 4);
    if (off > ws_size) return;  // workspace too small; nothing sane to do

    // 1) weights -> f16, zero BN stats
    prep_kernel<<<dim3((H0 * C0 + 255) / 256), 256, 0, stream>>>(w0, w1, w0h, w1h, st0, st1);
    // 2) points2 transpose
    transpose_p2_kernel<<<dim3(SS / 32, D2 / 32, BB), 256, 0, stream>>>(pts2, featsT);
    // 3) points1 -> x0 columns [0,128)
    p1_to_x0_kernel<<<dim3(NN / 32, D1 / 32, BB), 256, 0, stream>>>(pts1, x0h);
    // 4) 3-NN + weights
    knn_kernel<<<dim3(NN / 256, BB), 256, 0, stream>>>(xyz1, xyz2, idx4, w4);
    // 5) interpolation -> x0 columns [128,384)
    interp_kernel<<<dim3(MTOT), 128, 0, stream>>>(featsT, idx4, w4, x0h);
    // 6) layer 0 GEMM (WMMA f16 -> f32)
    gemm_wmma_kernel<C0, H0><<<dim3(MTOT / 128, H0 / 64), 256, 0, stream>>>(x0h, w0h, b0, y0);
    // 7) BN0 stats + apply + ReLU + f16 cast
    bn_stats_kernel<H0><<<dim3(MTOT / 256), 256, 0, stream>>>(y0, st0);
    bn_relu_cast_kernel<H0><<<dim3((size_t)MTOT * H0 / 256), 256, 0, stream>>>(y0, st0, g0, be0, x1h);
    // 8) layer 1 GEMM
    gemm_wmma_kernel<H0, H1><<<dim3(MTOT / 128, H1 / 64), 256, 0, stream>>>(x1h, w1h, b1, y1);
    // 9) BN1 stats + apply + ReLU + transpose to (B, 128, N)
    bn_stats_kernel<H1><<<dim3(MTOT / 256), 256, 0, stream>>>(y1, st1);
    bn_relu_out_kernel<<<dim3(NN / 32, H1 / 32, BB), 256, 0, stream>>>(y1, st1, g1, be1, out);

    (void)in_sizes; (void)n_in; (void)out_size;
}
